// SinkhornRouter_10402410791238
// MI455X (gfx1250) — matrix operand order repaired and checked by reading, not compiled, not run on hardware
//
#include <hip/hip_runtime.h>
#include <math.h>

typedef __attribute__((ext_vector_type(16))) __bf16 v16bf;
typedef __attribute__((ext_vector_type(8)))  float  v8f;

#define NTOK 8192
#define NEXP 64
#define HID  4096
#define GEMM_BLOCKS 128
#define GEMM_TPB 128          // 4 waves/block * 128 blocks = 512 M-tiles
#define SINK_BLOCKS 32
#define SINK_TPB 256
#define TOLV 1e-4f
#define EPSV 1e-8f
#define MAXIT 512

// Convert 8 fp32 (two float4) into hi/lo bf16 halves at element offset `base`
// of a v16bf pair (bf16x3 split-precision trick).
__device__ inline void cvt8(float4 a, float4 b, v16bf& hi, v16bf& lo, int base) {
  float v[8] = {a.x, a.y, a.z, a.w, b.x, b.y, b.z, b.w};
#pragma unroll
  for (int i = 0; i < 8; i++) {
    __bf16 h = (__bf16)v[i];
    hi[base + i] = h;
    lo[base + i] = (__bf16)(v[i] - (float)h);
  }
}

// ---------------------------------------------------------------------------
// Kernel 0: one-time split of W (64x4096 fp32, loop-invariant, 1 MB) into
// bf16-hi / bf16-lo planes so the GEMM hot loop loads bf16 directly.
// Also deterministically resets the grid-barrier state each call.
// ---------------------------------------------------------------------------
__global__ __launch_bounds__(256)
void wsplit_kernel(const float* __restrict__ W, __bf16* __restrict__ wh,
                   __bf16* __restrict__ wl, int* __restrict__ syncbuf) {
  if (blockIdx.x == 0 && threadIdx.x == 0) { syncbuf[0] = 0; syncbuf[1] = 0; }
  const int i = (blockIdx.x * 256 + threadIdx.x) * 4;   // 256 blocks cover 64*4096
  float4 v = *(const float4*)(W + i);
  float f[4] = {v.x, v.y, v.z, v.w};
#pragma unroll
  for (int t = 0; t < 4; t++) {
    __bf16 h = (__bf16)f[t];
    wh[i + t] = h;
    wl[i + t] = (__bf16)(f[t] - (float)h);
  }
}

// ---------------------------------------------------------------------------
// Kernel 1: logits = X @ W^T via bf16x3 WMMA (fp32-accurate).
// Wave-per-16-token tile, all 64 experts (4 N-tiles of 16), K step of 32.
// A layout (16x32 bf16): lanes 0-15 row M=l,   K = k0..k0+15 consecutive;
//                        lanes 16-31 row M=l-16, K = k0+16..k0+31.
// B layout (32x16 bf16): lanes 0-15 col N=l,   K = k0..k0+15 consecutive;
//                        lanes 16-31 col N=l-16, K = k0+16..k0+31.
// ---------------------------------------------------------------------------
__global__ __launch_bounds__(GEMM_TPB)
void router_gemm_kernel(const float* __restrict__ x,
                        const __bf16* __restrict__ wh, const __bf16* __restrict__ wl,
                        float* __restrict__ logits) {
  const int lane  = threadIdx.x & 31;
  const int wave  = threadIdx.x >> 5;
  const int tile  = blockIdx.x * (GEMM_TPB / 32) + wave;  // 512 tiles total
  const int mbase = tile * 16;
  const int lhalf = lane >> 4;                 // 0: lanes 0-15, 1: lanes 16-31
  const int l16   = lane & 15;

  const float* xr = x + (size_t)(mbase + l16) * HID;
  const int offA  = lhalf * 16;                // first A K-chunk
  const int offA2 = offA + 8;                  // second A K-chunk
  const int kB    = lhalf * 16;                // B consecutive K base

  v8f acc[4];
#pragma unroll
  for (int t = 0; t < 4; t++)
#pragma unroll
    for (int e = 0; e < 8; e++) acc[t][e] = 0.0f;

#pragma unroll 2
  for (int k = 0; k < HID; k += 32) {
    float4 a0 = *(const float4*)(xr + k + offA);
    float4 a1 = *(const float4*)(xr + k + offA + 4);
    float4 a2 = *(const float4*)(xr + k + offA2);
    float4 a3 = *(const float4*)(xr + k + offA2 + 4);
    v16bf ah, al;
    cvt8(a0, a1, ah, al, 0);
    cvt8(a2, a3, ah, al, 8);

#pragma unroll
    for (int nt = 0; nt < 4; nt++) {
      const size_t woff = (size_t)(nt * 16 + l16) * HID + k + kB;  // 32B aligned
      v16bf bh = *(const v16bf*)(wh + woff);
      v16bf bl = *(const v16bf*)(wl + woff);

      acc[nt] = __builtin_amdgcn_wmma_f32_16x16x32_bf16(false, ah, false, bh,
                                                        (short)0, acc[nt], false, false);
      acc[nt] = __builtin_amdgcn_wmma_f32_16x16x32_bf16(false, ah, false, bl,
                                                        (short)0, acc[nt], false, false);
      acc[nt] = __builtin_amdgcn_wmma_f32_16x16x32_bf16(false, al, false, bh,
                                                        (short)0, acc[nt], false, false);
    }
  }

  // C/D layout: lanes 0-15 -> rows 0..7 (VGPR r = row r), lanes 16-31 -> rows 8..15.
  const int mout = mbase + lhalf * 8;
#pragma unroll
  for (int nt = 0; nt < 4; nt++) {
    const int n = nt * 16 + l16;
#pragma unroll
    for (int r = 0; r < 8; r++) {
      logits[(size_t)(mout + r) * NEXP + n] = acc[nt][r];
    }
  }
}

// ---------------------------------------------------------------------------
// Kernel 2: device-resident Sinkhorn loop (cost rows live in registers),
// grid-cooperative with one barrier/iteration, fused top-2 + softmax gather.
// ---------------------------------------------------------------------------
__global__ __launch_bounds__(SINK_TPB)
void sinkhorn_topk_kernel(const float* __restrict__ logits, float* __restrict__ gpart,
                          int* __restrict__ syncbuf, float* __restrict__ out) {
  __shared__ float lds_part[8][64];
  __shared__ float d1s[64];
  __shared__ float errbuf[64];
  __shared__ float err_s;

  const int tid  = threadIdx.x;
  const int lane = tid & 31;
  const int wave = tid >> 5;
  const int row  = blockIdx.x * SINK_TPB + tid;

  // cost row c[j] = exp(logit[row][j]) kept in registers for the whole loop
  float c[64];
  const float* lr = logits + (size_t)row * NEXP;
#pragma unroll
  for (int q = 0; q < 16; q++) {
    float4 v = *(const float4*)(lr + q * 4);
    c[q * 4 + 0] = expf(v.x);
    c[q * 4 + 1] = expf(v.y);
    c[q * 4 + 2] = expf(v.z);
    c[q * 4 + 3] = expf(v.w);
  }

  if (tid < 64) d1s[tid] = 1.0f;
  __syncthreads();

  volatile int* gen = (volatile int*)(syncbuf + 1);
  int it = 0;
  float err;

  do {
    // d0_i = (1/N) / (sum_j d1_j * c_ij + eps)
    float s = 0.0f;
#pragma unroll
    for (int j = 0; j < 64; j++) s = fmaf(d1s[j], c[j], s);
    const float d0 = (1.0f / (float)NTOK) / (s + EPSV);

    // per-thread weighted row, butterfly-reduced across the wave
    float p[64];
#pragma unroll
    for (int j = 0; j < 64; j++) p[j] = d0 * c[j];
#pragma unroll
    for (int off = 16; off > 0; off >>= 1) {
#pragma unroll
      for (int j = 0; j < 64; j++) p[j] += __shfl_xor(p[j], off, 32);
    }
    if (lane == 0) {
#pragma unroll
      for (int j = 0; j < 64; j++) lds_part[wave][j] = p[j];
    }
    __syncthreads();

    const int parity = it & 1;
    float* gp = gpart + (size_t)parity * (SINK_BLOCKS * 64);
    if (tid < 64) {
      float v = 0.0f;
#pragma unroll
      for (int w = 0; w < 8; w++) v += lds_part[w][tid];
      gp[blockIdx.x * 64 + tid] = v;
      __threadfence();
    }

    // ---- grid barrier (arrive counter + generation spin) ----
    __syncthreads();
    if (tid == 0) {
      const int g = *gen;
      __threadfence();
      if (atomicAdd(syncbuf, 1) == SINK_BLOCKS - 1) {
        syncbuf[0] = 0;
        __threadfence();
        atomicAdd(syncbuf + 1, 1);
      } else {
        while (*gen == g) __builtin_amdgcn_s_sleep(1);
      }
    }
    __syncthreads();

    // every block redundantly computes the identical d1_new and err
    if (tid < 64) {
      float col = 0.0f;
#pragma unroll
      for (int b = 0; b < SINK_BLOCKS; b++) col += gp[b * 64 + tid];
      const float d1n = (1.0f / (float)NEXP) / (col + EPSV);
      errbuf[tid] = fabsf(d1s[tid] - d1n);
      d1s[tid] = d1n;
    }
    __syncthreads();
    if (tid == 0) {
      float e = 0.0f;
#pragma unroll
      for (int j = 0; j < 64; j++) e += errbuf[j];
      err_s = e * (1.0f / 64.0f);
    }
    __syncthreads();
    err = err_s;
    ++it;
  } while (err > TOLV && it < MAXIT);

  // top-2 over d1_j * c_ij (d0_i > 0 cannot change per-row ordering),
  // scores from plain softmax: c[j] / sum_j c[j]
  float sum = 0.0f;
#pragma unroll
  for (int j = 0; j < 64; j++) sum += c[j];

  float best = -__builtin_inff(), second = -__builtin_inff();
  float cb = 0.0f, cs = 0.0f;
  int bi = 0, si = 0;
#pragma unroll
  for (int j = 0; j < 64; j++) {
    const float key = d1s[j] * c[j];
    if (key > best) {
      second = best; si = bi; cs = cb;
      best = key;    bi = j;  cb = c[j];
    } else if (key > second) {
      second = key;  si = j;  cs = c[j];
    }
  }

  const float inv = 1.0f / sum;
  out[(size_t)row * 2 + 0] = cb * inv;
  out[(size_t)row * 2 + 1] = cs * inv;
  out[(size_t)NTOK * 2 + (size_t)row * 2 + 0] = (float)bi;
  out[(size_t)NTOK * 2 + (size_t)row * 2 + 1] = (float)si;
}

// ---------------------------------------------------------------------------
extern "C" void kernel_launch(void* const* d_in, const int* in_sizes, int n_in,
                              void* d_out, int out_size, void* d_ws, size_t ws_size,
                              hipStream_t stream) {
  const float* x = (const float*)d_in[0];   // (2048, 4, 4096) fp32 -> (8192, 4096)
  const float* W = (const float*)d_in[1];   // (64, 4096) fp32
  float* out = (float*)d_out;               // scores (8192,2) then indices (8192,2)

  float*  logits  = (float*)d_ws;                           // 8192*64 fp32 = 2 MB
  float*  gpart   = logits + (size_t)NTOK * NEXP;           // 2*32*64 fp32 (double buffer)
  int*    syncbuf = (int*)(gpart + 2 * SINK_BLOCKS * NEXP); // {counter, generation}
  __bf16* wh      = (__bf16*)(syncbuf + 4);                 // 64*4096 bf16 (hi)
  __bf16* wl      = wh + (size_t)NEXP * HID;                // 64*4096 bf16 (lo)

  wsplit_kernel<<<(NEXP * HID) / (256 * 4), 256, 0, stream>>>(W, wh, wl, syncbuf);
  router_gemm_kernel<<<GEMM_BLOCKS, GEMM_TPB, 0, stream>>>(x, wh, wl, logits);
  sinkhorn_topk_kernel<<<SINK_BLOCKS, SINK_TPB, 0, stream>>>(logits, gpart, syncbuf, out);
}
